// Attn_FullLevel_84945863180982
// MI455X (gfx1250) — matrix operand order repaired and checked by reading, not compiled, not run on hardware
//
#include <hip/hip_runtime.h>

// ---------------------------------------------------------------------------
// MI455X (gfx1250) fused two-level attention.
// B=32, V=64, N=64, D=256, P=32. All GEMMs via v_wmma_f32_16x16x32_bf16.
// Memory-bound problem (~86 GFLOP vs 512MB min traffic @ 23.3TB/s):
// strategy = bf16 intermediates, minimal passes, LDS-fused softmax,
// ASYNCcnt-tracked global->LDS staging where no transpose/convert is needed.
// ---------------------------------------------------------------------------

#define BB 32
#define VV 64
#define NN 64
#define DD 256
#define PP 32

typedef __attribute__((ext_vector_type(16))) __bf16 v16bf;
typedef __attribute__((ext_vector_type(8)))  __bf16 v8bf;
typedef __attribute__((ext_vector_type(8)))  float  v8f;
typedef __attribute__((ext_vector_type(4)))  int    v4i;

// ---- async global->LDS (GLOBAL_LOAD_ASYNC_TO_LDS_B128, ASYNCcnt) ----------
// clang prototype (from round-2 diagnostic): arg0 = v4i addrspace(1)* (global
// source, printed as "int __vector(4) __device__*"), arg1 = LDS side, then
// two immediates (offset, cpol).
#if __has_builtin(__builtin_amdgcn_global_load_async_to_lds_b128) && \
    __has_builtin(__builtin_amdgcn_s_wait_asynccnt)
#define USE_ASYNC_LDS 1
typedef __attribute__((address_space(1))) v4i* gas_ptr;
typedef __attribute__((address_space(3))) v4i* las_ptr;
__device__ __forceinline__ void async_cp16(void* lds_dst, const void* gsrc) {
  __builtin_amdgcn_global_load_async_to_lds_b128((gas_ptr)gsrc, (las_ptr)lds_dst,
                                                 /*offset=*/0, /*cpol=*/0);
}
__device__ __forceinline__ void async_wait0() {
  __builtin_amdgcn_s_wait_asynccnt(0);
}
#else
#define USE_ASYNC_LDS 0
#endif

// ---- WMMA fragment helpers (ISA 7.12.2 layouts, wave32) -------------------
// A 16x32 bf16: lane L<16 holds row M=L, K = {0..7,16..23}; lanes 16..31 hold
// the same rows with K = {8..15,24..31}. Caller pre-adds base = (lane>>4)*8.
__device__ __forceinline__ v16bf frag_a_ld(const __bf16* p) {
  union { v16bf v; v8bf h[2]; } u;
  u.h[0] = *(const v8bf*)(p);        // K base+0 .. base+7
  u.h[1] = *(const v8bf*)(p + 16);   // K base+16 .. base+23
  return u.v;
}
__device__ __forceinline__ v16bf frag_a_ld(const float* p) {
  v16bf r;
#pragma unroll
  for (int i = 0; i < 8; ++i) r[i] = (__bf16)p[i];
#pragma unroll
  for (int i = 0; i < 8; ++i) r[i + 8] = (__bf16)p[i + 16];
  return r;
}
// B 32x16 bf16: lane L<16 holds column N=L, K = {0..7,16..23}; lanes 16..31
// hold the same columns, K = {8..15,24..31}. Caller pre-adds koff = (lane>>4)*8.
__device__ __forceinline__ v16bf frag_b_ld(const __bf16* p) {
  union { v16bf v; v8bf h[2]; } u;
  u.h[0] = *(const v8bf*)(p);
  u.h[1] = *(const v8bf*)(p + 16);
  return u.v;
}
__device__ __forceinline__ v8f wmma_bf16(v16bf a, v16bf b, v8f c) {
  return __builtin_amdgcn_wmma_f32_16x16x32_bf16(
      /*neg_a=*/false, a, /*neg_b=*/false, b,
      /*c_mod=*/(short)0, c, /*reuse_a=*/false, /*reuse_b=*/false);
}

// ---------------------------------------------------------------------------
// proj_gemm: Y[map_out(r), :] = X[map_in(r), :] @ W(256x256) + bias
// map(r; rpg, stride, off) = (r/rpg)*stride + off + r%rpg   (row remapping so
// the same kernel serves full tensors, last-P slices, and output splicing).
// Block = 256 threads (8 waves), tile = 128 rows x 256 cols, K = 256.
// W is staged transposed+converted in LDS (B-frag reads contiguous); A-frags
// load directly from global (each reused across all 16 N-tiles in registers).
// ---------------------------------------------------------------------------
template <typename InT, typename OutT>
__global__ void proj_gemm(const InT* __restrict__ X,
                          const float* __restrict__ Wm,
                          const float* __restrict__ bias,
                          OutT* __restrict__ Y,
                          int in_rpg, int in_stride, int in_off,
                          int out_rpg, int out_stride, int out_off) {
  extern __shared__ char smem_raw[];
  __bf16* sWt = (__bf16*)smem_raw;         // [256][LDW] transposed weights
  const int LDW = 272;                     // pad: rows 32B-aligned, bank-spread

  for (int i = threadIdx.x; i < 256 * 256; i += 256) {
    int k = i >> 8, n = i & 255;
    sWt[n * LDW + k] = (__bf16)Wm[i];
  }
  __syncthreads();

  const int wave = threadIdx.x >> 5, lane = threadIdx.x & 31;
  const long tile_base = (long)blockIdx.x * 128;
  const int m0 = wave * 16;

  // A fragments for this wave's 16 rows, all 8 K-steps.
  v16bf aF[8];
  {
    long lr = tile_base + m0 + (lane & 15);
    long g = (lr / in_rpg) * (long)in_stride + in_off + (lr % in_rpg);
    const InT* rowp = X + g * DD + ((lane >> 4) << 3);
#pragma unroll
    for (int kt = 0; kt < 8; ++kt) aF[kt] = frag_a_ld(rowp + kt * 32);
  }

#pragma unroll 4
  for (int nt = 0; nt < 16; ++nt) {
    const int n0 = nt * 16;
    const __bf16* bp = sWt + (n0 + (lane & 15)) * LDW + ((lane >> 4) << 3);
    v8f acc = {};
#pragma unroll
    for (int kt = 0; kt < 8; ++kt)
      acc = wmma_bf16(aF[kt], frag_b_ld(bp + kt * 32), acc);

    const float bv = bias[n0 + (lane & 15)];
    const int col = n0 + (lane & 15);
    const int rb = m0 + ((lane >> 4) << 3);
#pragma unroll
    for (int r = 0; r < 8; ++r) {
      long lr = tile_base + rb + r;
      long g = (lr / out_rpg) * (long)out_stride + out_off + (lr % out_rpg);
      Y[g * DD + col] = (OutT)(acc[r] + bv);
    }
  }
}

// ---------------------------------------------------------------------------
// scoresV: per (b,p): S[v,w] = sum_d q[b,v,p,d] * k2[b,w,p,d]   (64x256x64 NT)
// Both operands contiguous along K=d in the compact (B,V,P,D) buffers,
// so every fragment loads straight from global. Output (B,V,P,W) f32.
// ---------------------------------------------------------------------------
__global__ void scoresV_kernel(const __bf16* __restrict__ qb,
                               const __bf16* __restrict__ k2b,
                               float* __restrict__ scV) {
  const int b = blockIdx.x >> 5, p = blockIdx.x & 31;
  const int wave = threadIdx.x >> 5, lane = threadIdx.x & 31;
  const int mt = wave >> 1, m0 = mt * 16;

  const __bf16* arow =
      qb + ((size_t)(b * VV + m0 + (lane & 15)) * PP + p) * DD + ((lane >> 4) << 3);
  v16bf aF[8];
#pragma unroll
  for (int kt = 0; kt < 8; ++kt) aF[kt] = frag_a_ld(arow + kt * 32);

#pragma unroll
  for (int j = 0; j < 2; ++j) {
    const int n0 = ((wave & 1) * 2 + j) * 16;
    const __bf16* brow =
        k2b + ((size_t)(b * VV + n0 + (lane & 15)) * PP + p) * DD + ((lane >> 4) << 3);
    v8f acc = {};
#pragma unroll
    for (int kt = 0; kt < 8; ++kt)
      acc = wmma_bf16(aF[kt], frag_b_ld(brow + kt * 32), acc);

    const int col = n0 + (lane & 15);
    const int rb = m0 + ((lane >> 4) << 3);
#pragma unroll
    for (int r = 0; r < 8; ++r)
      scV[((size_t)(b * VV + rb + r) * PP + p) * VV + col] = acc[r];
  }
}

// ---------------------------------------------------------------------------
// attn_fused: per (b,v):
//   scoresT = qP @ k1^T (WMMA -> LDS f32 cols 0..63)
//   append scoresV slice (cols 64..127), softmax rows of 128 (shfl_xor)
//   attnT -> LDS bf16 (WMMA A-operand), attnV -> global bf16
//   outT  = attnT @ v1 (v1 staged transposed in LDS) -> f32 workspace
// k1 slab and scoresV slice are staged via GLOBAL_LOAD_ASYNC_TO_LDS_B128
// (ASYNCcnt) when available: no VGPR round-trip for bulk row-major staging.
// ---------------------------------------------------------------------------
__global__ void attn_fused(const __bf16* __restrict__ qb,
                           const __bf16* __restrict__ k1b,
                           const __bf16* __restrict__ v1b,
                           const float* __restrict__ scV,
                           __bf16* __restrict__ attnVb,
                           float* __restrict__ outTb) {
  extern __shared__ char smem_raw[];
  const int LK = 272, LV = 80, LS = 128, LA = 72;
  __bf16* sK1 = (__bf16*)smem_raw;                 // [64][LK]   34816 B
  __bf16* sV1t = sK1 + 64 * LK;                    // [256][LV]  40960 B
  float* sSc = (float*)(sV1t + 256 * LV);          // [32][128]  16384 B
  __bf16* sAT = (__bf16*)(sSc + 32 * LS);          // [32][LA]    4608 B

  const int bv = blockIdx.x;
  const size_t kvbase = (size_t)bv * NN * DD;
  const size_t qbase = (size_t)bv * PP * DD;

#if USE_ASYNC_LDS
  // stage k1 (row-major): 2048 x 16B async chunks, 8 per thread
  for (int c = threadIdx.x; c < (NN * DD) / 8; c += 256) {
    int s = c >> 5, d0 = (c & 31) << 3;
    async_cp16(sK1 + s * LK + d0, k1b + kvbase + s * DD + d0);
  }
  // stage scoresV slice into cols 64..127: 512 x 16B async chunks
  for (int c = threadIdx.x; c < (PP * VV) / 4; c += 256) {
    int p = c >> 4, w0 = (c & 15) << 2;
    async_cp16(sSc + p * LS + 64 + w0, scV + (size_t)bv * PP * VV + p * VV + w0);
  }
#else
  for (int c = threadIdx.x; c < (NN * DD) / 8; c += 256) {
    int s = c >> 5, d0 = (c & 31) << 3;
    *(uint4*)(sK1 + s * LK + d0) = *(const uint4*)(k1b + kvbase + s * DD + d0);
  }
  for (int c = threadIdx.x; c < (PP * VV) / 4; c += 256) {
    int p = c >> 4, w0 = (c & 15) << 2;
    *(float4*)(sSc + p * LS + 64 + w0) =
        *(const float4*)(scV + (size_t)bv * PP * VV + p * VV + w0);
  }
#endif
  // stage v1 transposed: sV1t[d][s] (element transpose -> stays on sync path)
  for (int i = threadIdx.x; i < NN * DD; i += 256) {
    int s = i >> 8, d = i & 255;
    sV1t[d * LV + s] = v1b[kvbase + i];
  }
#if USE_ASYNC_LDS
  async_wait0();
#endif
  __syncthreads();

  const int wave = threadIdx.x >> 5, lane = threadIdx.x & 31;

  // ---- scoresT: 2 M-tiles x 4 N-tiles, one per wave -----------------------
  {
    const int m0 = (wave >> 2) * 16, n0 = (wave & 3) * 16;
    const __bf16* arow =
        qb + qbase + (size_t)(m0 + (lane & 15)) * DD + ((lane >> 4) << 3);
    const __bf16* bp = sK1 + (n0 + (lane & 15)) * LK + ((lane >> 4) << 3);
    v8f acc = {};
#pragma unroll
    for (int kt = 0; kt < 8; ++kt)
      acc = wmma_bf16(frag_a_ld(arow + kt * 32), frag_b_ld(bp + kt * 32), acc);
    const int col = n0 + (lane & 15);
    const int rb = m0 + ((lane >> 4) << 3);
#pragma unroll
    for (int r = 0; r < 8; ++r) sSc[(rb + r) * LS + col] = acc[r];
  }
  __syncthreads();

  // ---- softmax over 128-wide rows; wave handles 4 rows --------------------
  const float scale = 0.0625f;  // 1/sqrt(256)
#pragma unroll
  for (int i = 0; i < 4; ++i) {
    const int row = wave * 4 + i;
    float x[4];
#pragma unroll
    for (int j = 0; j < 4; ++j) x[j] = sSc[row * LS + lane + 32 * j] * scale;
    float m = fmaxf(fmaxf(x[0], x[1]), fmaxf(x[2], x[3]));
#pragma unroll
    for (int off = 16; off; off >>= 1) m = fmaxf(m, __shfl_xor(m, off, 32));
    float s = 0.f;
#pragma unroll
    for (int j = 0; j < 4; ++j) { x[j] = __expf(x[j] - m); s += x[j]; }
#pragma unroll
    for (int off = 16; off; off >>= 1) s += __shfl_xor(s, off, 32);
    const float inv = 1.0f / s;
#pragma unroll
    for (int j = 0; j < 4; ++j) {
      const int c = lane + 32 * j;
      const float a = x[j] * inv;
      if (j < 2)  // attnT cols 0..63 -> LDS (A operand of outT)
        sAT[row * LA + c] = (__bf16)a;
      else        // attnV cols 64..127 -> global (consumed by outV per (b,p))
        attnVb[(size_t)bv * PP * VV + row * VV + (c - 64)] = (__bf16)a;
    }
  }
  __syncthreads();

  // ---- outT = attnT(32x64) @ v1(64x256): 2x16 tiles, 4 per wave -----------
  {
    const int m0 = (wave & 1) * 16;
    const __bf16* arow = sAT + (m0 + (lane & 15)) * LA + ((lane >> 4) << 3);
    v16bf aF[2] = { frag_a_ld(arow), frag_a_ld(arow + 32) };
#pragma unroll
    for (int j = 0; j < 4; ++j) {
      const int n0 = ((wave >> 1) * 4 + j) * 16;
      const __bf16* bp = sV1t + (n0 + (lane & 15)) * LV + ((lane >> 4) << 3);
      v8f acc = {};
#pragma unroll
      for (int kt = 0; kt < 2; ++kt)
        acc = wmma_bf16(aF[kt], frag_b_ld(bp + kt * 32), acc);
      const int col = n0 + (lane & 15);
      const int rb = m0 + ((lane >> 4) << 3);
#pragma unroll
      for (int r = 0; r < 8; ++r)
        outTb[(size_t)bv * PP * DD + (size_t)(rb + r) * DD + col] = acc[r];
    }
  }
}

// ---------------------------------------------------------------------------
// outV: per (b,p): O[v,d] = sum_w attnV[b,v,p,w] * v2[b,w,p,d]; outCV = O+outT
// attnV rows are contiguous along K=w (direct global A-frags); v2P staged
// transposed in LDS for the B operand. Emits bf16 outCV (B,V,P,D).
// ---------------------------------------------------------------------------
__global__ void outV_kernel(const __bf16* __restrict__ attnVb,
                            const __bf16* __restrict__ v2b,
                            const float* __restrict__ outTb,
                            __bf16* __restrict__ outCVb) {
  extern __shared__ char smem_raw[];
  __bf16* sV2t = (__bf16*)smem_raw;  // [256][LV]
  const int LV = 80;
  const int b = blockIdx.x >> 5, p = blockIdx.x & 31;

  for (int i = threadIdx.x; i < VV * DD; i += 256) {
    int w = i >> 8, d = i & 255;
    sV2t[d * LV + w] = v2b[((size_t)(b * VV + w) * PP + p) * DD + d];
  }
  __syncthreads();

  const int wave = threadIdx.x >> 5, lane = threadIdx.x & 31;
  const int m0 = (wave >> 1) * 16;
  const __bf16* arow =
      attnVb + ((size_t)(b * VV + m0 + (lane & 15)) * PP + p) * VV + ((lane >> 4) << 3);
  v16bf aF[2] = { frag_a_ld(arow), frag_a_ld(arow + 32) };

#pragma unroll
  for (int j = 0; j < 8; ++j) {
    const int n0 = ((wave & 1) * 8 + j) * 16;
    const __bf16* bp = sV2t + (n0 + (lane & 15)) * LV + ((lane >> 4) << 3);
    v8f acc = {};
#pragma unroll
    for (int kt = 0; kt < 2; ++kt)
      acc = wmma_bf16(aF[kt], frag_b_ld(bp + kt * 32), acc);
    const int col = n0 + (lane & 15);
    const int rb = m0 + ((lane >> 4) << 3);
#pragma unroll
    for (int r = 0; r < 8; ++r) {
      const size_t idx = ((size_t)(b * VV + rb + r) * PP + p) * DD + col;
      outCVb[idx] = (__bf16)(acc[r] + outTb[idx]);
    }
  }
}

// ---------------------------------------------------------------------------
extern "C" void kernel_launch(void* const* d_in, const int* in_sizes, int n_in,
                              void* d_out, int out_size, void* d_ws, size_t ws_size,
                              hipStream_t stream) {
  (void)in_sizes; (void)n_in; (void)out_size; (void)ws_size;
  const float* queries = (const float*)d_in[0];
  const float* keys    = (const float*)d_in[1];
  const float* values  = (const float*)d_in[2];
  const float* Wq  = (const float*)d_in[3];  const float* bq  = (const float*)d_in[4];
  const float* Wk1 = (const float*)d_in[5];  const float* bk1 = (const float*)d_in[6];
  const float* Wk2 = (const float*)d_in[7];  const float* bk2 = (const float*)d_in[8];
  const float* Wv1 = (const float*)d_in[9];  const float* bv1 = (const float*)d_in[10];
  const float* Wv2 = (const float*)d_in[11]; const float* bv2 = (const float*)d_in[12];
  const float* Wo  = (const float*)d_in[13]; const float* bo  = (const float*)d_in[14];
  float* out = (float*)d_out;

  // workspace layout (bytes)
  char* ws = (char*)d_ws;
  __bf16* qbuf   = (__bf16*)(ws + 0);          //  33,554,432  (B,V,P,D)
  __bf16* k1buf  = (__bf16*)(ws + 33554432);   //  67,108,864  (B,V,N,D)
  __bf16* v1buf  = (__bf16*)(ws + 100663296);  //  67,108,864  (B,V,N,D)
  __bf16* k2buf  = (__bf16*)(ws + 167772160);  //  33,554,432  (B,V,P,D)
  __bf16* v2buf  = (__bf16*)(ws + 201326592);  //  33,554,432  (B,V,P,D)
  float*  scV    = (float*)(ws + 234881024);   //  16,777,216  (B,V,P,W)
  __bf16* attnV  = (__bf16*)(ws + 251658240);  //   8,388,608  (B,V,P,W)
  float*  outT   = (float*)(ws + 260046848);   //  67,108,864  (B,V,P,D)
  __bf16* outCV  = (__bf16*)(ws + 327155712);  //  33,554,432  (B,V,P,D)

  const dim3 blk(256);
  const size_t ldsProj = 256 * 272 * sizeof(__bf16);                    // 139,264
  const size_t ldsFuse = (64 * 272 + 256 * 80 + 32 * 72) * 2 + 32 * 128 * 4; // 96,768
  const size_t ldsOutV = 256 * 80 * sizeof(__bf16);                     //  40,960

  const int M_full = BB * VV * NN;  // 131072 rows -> 1024 tiles
  const int M_last = BB * VV * PP;  //  65536 rows ->  512 tiles

  // projections (q/k2/v2 computed only on the last P rows of each (b,v))
  proj_gemm<float, __bf16><<<M_last / 128, blk, ldsProj, stream>>>(
      queries, Wq, bq, qbuf, PP, NN, NN - PP, PP, PP, 0);
  proj_gemm<float, __bf16><<<M_full / 128, blk, ldsProj, stream>>>(
      keys, Wk1, bk1, k1buf, NN, NN, 0, NN, NN, 0);
  proj_gemm<float, __bf16><<<M_full / 128, blk, ldsProj, stream>>>(
      values, Wv1, bv1, v1buf, NN, NN, 0, NN, NN, 0);
  proj_gemm<float, __bf16><<<M_last / 128, blk, ldsProj, stream>>>(
      keys, Wk2, bk2, k2buf, PP, NN, NN - PP, PP, PP, 0);
  proj_gemm<float, __bf16><<<M_last / 128, blk, ldsProj, stream>>>(
      values, Wv2, bv2, v2buf, PP, NN, NN - PP, PP, PP, 0);

  // cross-v scores, fused softmax+outT, cross-v output
  scoresV_kernel<<<BB * PP, blk, 0, stream>>>(qbuf, k2buf, scV);
  attn_fused<<<BB * VV, blk, ldsFuse, stream>>>(qbuf, k1buf, v1buf, scV, attnV, outT);
  outV_kernel<<<BB * PP, blk, ldsOutV, stream>>>(attnV, v2buf, outT, outCV);

  // final projection: rows [0,N-P) come from v1, rows [N-P,N) from outCV
  proj_gemm<__bf16, float><<<M_last / 128, blk, ldsProj, stream>>>(
      v1buf, Wo, bo, out, NN - PP, NN, 0, NN - PP, NN, 0);
  proj_gemm<__bf16, float><<<M_last / 128, blk, ldsProj, stream>>>(
      outCV, Wo, bo, out, PP, PP, 0, PP, NN, NN - PP);
}
